// Quantizer_11038065950768
// MI455X (gfx1250) — matrix operand order repaired and checked by reading, not compiled, not run on hardware
//
#include <hip/hip_runtime.h>

typedef __attribute__((ext_vector_type(16))) _Float16 v16h;
typedef __attribute__((ext_vector_type(8)))  _Float16 v8h;
typedef __attribute__((ext_vector_type(8)))  float    v8f;

#define BT     32768
#define CIN    512
#define NE     640      // GROUP*ENTRY
#define NGRP   2
#define ENTRY  320
#define DGRP   256
#define COUT   512
// GEMM1 tiling: Ktiles = 512/32 = 16, Ntiles = 640/16 = 40 (10 per wave)
// GEMM2 tiling per group: Ktiles = 320/32 = 10, Ntiles = 256/16 = 16

// ---- Pre-swizzle Wp [640,512] f32 -> f16 B-fragments -------------------
// B-fragment layout (V_WMMA_F32_16X16X32_F16, wave32, ISA 7.12.2):
//   lane L: n = L%16, lane-group g = L/16 ; half i (0..15): k = (i/8)*16 + g*8 + i%8
// Fragment id f = nt*16 + kt ; dst[(f*32 + lane)*16 + i]
__global__ void swizzle_wp(const float* __restrict__ Wp, _Float16* __restrict__ dst) {
    int tid = blockIdx.x * 256 + threadIdx.x;
    if (tid >= NE * CIN) return;
    int i    = tid & 15;
    int lane = (tid >> 4) & 31;
    int f    = tid >> 9;
    int kt   = f & 15;
    int nt   = f >> 4;                       // 0..39
    int n = nt * 16 + (lane & 15);           // output channel e
    int k = kt * 32 + (i >> 3) * 16 + (lane >> 4) * 8 + (i & 7);  // input channel c
    dst[tid] = (_Float16)Wp[(size_t)n * CIN + k];
}

// ---- Pre-swizzle codebooks [2,320,256] f32 -> f16 B-fragments ----------
// Fragment id f = (g*16 + nt)*10 + kt
__global__ void swizzle_cb(const float* __restrict__ cb, _Float16* __restrict__ dst) {
    int tid = blockIdx.x * 256 + threadIdx.x;
    if (tid >= NGRP * ENTRY * DGRP) return;
    int i    = tid & 15;
    int lane = (tid >> 4) & 31;
    int f    = tid >> 9;                     // 0..319
    int kt   = f % 10;
    int rest = f / 10;
    int nt   = rest & 15;
    int g    = rest >> 4;
    int n = nt * 16 + (lane & 15);           // d
    int k = kt * 32 + (i >> 3) * 16 + (lane >> 4) * 8 + (i & 7);  // e within group
    dst[tid] = (_Float16)cb[((size_t)g * ENTRY + k) * DGRP + n];
}

// Build A-fragment (16x32 f16, ISA 7.12.2): lane L row m=L%16, group g=L/16,
// halves 0..7 -> k = g*8+0..7, halves 8..15 -> k = 16+g*8+0..7  (two b128 DS loads)
__device__ __forceinline__ v16h load_a_frag(const _Float16* p /* row base + ktile*32 + g*8 */) {
    v8h lo = *(const v8h*)(p);
    v8h hi = *(const v8h*)(p + 16);
    return __builtin_shufflevector(lo, hi, 0,1,2,3,4,5,6,7,8,9,10,11,12,13,14,15);
}

// Async DMA: global -> LDS, 16B per lane, tracked by ASYNCcnt (not LOADcnt),
// so it overlaps the GEMM's s_wait_loadcnt-paced pipeline.
__device__ __forceinline__ void async_copy_b128(unsigned lds_byte_off,
                                                const void* sbase,
                                                unsigned global_byte_off) {
    asm volatile("global_load_async_to_lds_b128 %0, %1, %2"
                 :: "v"(lds_byte_off), "v"(global_byte_off), "s"(sbase)
                 : "memory");
}
__device__ __forceinline__ void wait_asynccnt0() {
    asm volatile("s_wait_asynccnt 0x0" ::: "memory");
}

__launch_bounds__(128)
__global__ void quantizer_fused(const float* __restrict__ x,
                                const float* __restrict__ bp,
                                const float* __restrict__ gumbel,
                                const _Float16* __restrict__ wpf,
                                const _Float16* __restrict__ cbf,
                                float* __restrict__ out) {
    // 40KB logits (aliases 16KB x-f16 staging) + 20KB y-f16 + 40KB gumbel = 100KB
    __shared__ __align__(16) float     s_logits[16 * NE];
    __shared__ __align__(16) _Float16  s_y[16 * NE];
    __shared__ __align__(16) float     s_gum[16 * NE];
    _Float16* s_x = (_Float16*)s_logits;     // [16][512] f16, reused before logits exist

    const int tid  = threadIdx.x;
    const int lane = tid & 31;
    const int wave = tid >> 5;               // 0..3
    const int lg   = lane >> 4;              // lane group (0/1)
    const int ln   = lane & 15;              // n col (B/C/D) == m row (A)
    const int row0 = blockIdx.x * 16;

    // ---- Phase 0a: kick off async DMA of the gumbel tile (overlaps GEMM1) ----
    // 16 rows x 640 f32 = 2560 x 16B chunks; row-major both sides.
    {
        const unsigned lds0 = (unsigned)(uintptr_t)s_gum;
        for (int idx = tid; idx < 16 * (NE / 4); idx += 128) {   // NE/4 f32x4 per row
            const int m = idx / (NE / 4);
            const int c = idx % (NE / 4);
            async_copy_b128(lds0 + (unsigned)idx * 16u,
                            gumbel,
                            (unsigned)((row0 + m) * (NE * 4) + c * 16));
        }
    }

    // ---- Phase 0b: stage x tile to LDS as f16 (coalesced) ----
    for (int idx = tid; idx < 16 * CIN; idx += 128) {
        s_x[idx] = (_Float16)x[(size_t)(row0 + (idx >> 9)) * CIN + (idx & 511)];
    }
    __syncthreads();

    // ---- Phase 1: logits = x @ Wp^T ; each wave owns 10 N-tiles ----
    // Software pipeline: prefetch next B (and next A at tile edge) while the
    // current WMMA runs, so s_wait_loadcnt finds data already resident.
    v8f acc[10];
#pragma unroll
    for (int t = 0; t < 10; ++t) acc[t] = (v8f){};

    // wave-constant bases: all fragment offsets become compile-time immediates
    const _Float16* wbase = wpf + ((size_t)(wave * 160) * 32 + lane) * 16;
    const _Float16* abase = s_x + ln * CIN + lg * 8;

    v16h a_cur = load_a_frag(abase);
    v16h b_cur = *(const v16h*)(wbase);
#pragma unroll
    for (int kt = 0; kt < 16; ++kt) {
        v16h a_nxt = a_cur;
        if (kt < 15) a_nxt = load_a_frag(abase + (kt + 1) * 32);
#pragma unroll
        for (int t = 0; t < 10; ++t) {
            v16h b_nxt = b_cur;
            if (!(kt == 15 && t == 9)) {
                const int t2  = (t == 9) ? 0 : t + 1;
                const int kt2 = (t == 9) ? kt + 1 : kt;
                b_nxt = *(const v16h*)(wbase + (size_t)(t2 * 16 + kt2) * 512);
            }
            acc[t] = __builtin_amdgcn_wmma_f32_16x16x32_f16(
                false, a_cur, false, b_cur, (short)0, acc[t], false, false);
            b_cur = b_nxt;
        }
        a_cur = a_nxt;
    }
    __syncthreads();   // all waves done reading s_x; safe to overwrite with logits
#pragma unroll
    for (int t = 0; t < 10; ++t) {
        const int ntg = wave * 10 + t;
#pragma unroll
        for (int r = 0; r < 8; ++r)          // D layout: m = r + 8*lg, n = ln
            s_logits[(r + 8 * lg) * NE + ntg * 16 + ln] = acc[t][r];
    }
    wait_asynccnt0();                        // this wave's gumbel DMA done
    __syncthreads();                         // => whole tile's DMA visible

    // ---- Phase 2: softmax(logits + bias + gumbel) over 640; 8 lanes/row ----
    {
        const int row = tid >> 3;            // 0..15
        const int j   = tid & 7;
        float* lrow = s_logits + row * NE;
        const float* grow = s_gum + row * NE;
        float m = -3.4e38f;
        for (int e = j; e < NE; e += 8) {
            float v = lrow[e] + bp[e] + grow[e];
            lrow[e] = v;
            m = fmaxf(m, v);
        }
        m = fmaxf(m, __shfl_xor(m, 1));
        m = fmaxf(m, __shfl_xor(m, 2));
        m = fmaxf(m, __shfl_xor(m, 4));
        float s = 0.0f;
        for (int e = j; e < NE; e += 8) {
            float ex = __expf(lrow[e] - m);
            lrow[e] = ex;
            s += ex;
        }
        s += __shfl_xor(s, 1);
        s += __shfl_xor(s, 2);
        s += __shfl_xor(s, 4);
        const float inv = 1.0f / s;
        for (int e = j; e < NE; e += 8)
            s_y[row * NE + e] = (_Float16)(lrow[e] * inv);
    }
    __syncthreads();

    // ---- Phase 3: out[:, g*256+d] = y_g @ codebook_g ; each wave owns 8 N-tiles ----
    const _Float16* cbase = cbf + ((size_t)(wave * 80) * 32 + lane) * 16;
#pragma unroll
    for (int jq = 0; jq < 8; ++jq) {
        const int q  = wave * 8 + jq;        // 0..31
        const int g  = q >> 4;
        const int nt = q & 15;
        const _Float16* ybase = s_y + ln * NE + g * ENTRY + lg * 8;
        v8f c = (v8f){};
        v16h a2 = load_a_frag(ybase);
        v16h b2 = *(const v16h*)(cbase + (size_t)(jq * 10) * 512);
#pragma unroll
        for (int kt = 0; kt < 10; ++kt) {
            v16h a_nxt = a2;
            v16h b_nxt = b2;
            if (kt < 9) {
                a_nxt = load_a_frag(ybase + (kt + 1) * 32);
                b_nxt = *(const v16h*)(cbase + (size_t)(jq * 10 + kt + 1) * 512);
            } else if (jq < 7) {
                // prefetch first fragments of the next N-tile
                const int q2  = wave * 8 + jq + 1;
                const int g2  = q2 >> 4;
                a_nxt = load_a_frag(s_y + ln * NE + g2 * ENTRY + lg * 8);
                b_nxt = *(const v16h*)(cbase + (size_t)((jq + 1) * 10) * 512);
            }
            c = __builtin_amdgcn_wmma_f32_16x16x32_f16(
                false, a2, false, b2, (short)0, c, false, false);
            a2 = a_nxt;
            b2 = b_nxt;
        }
#pragma unroll
        for (int r = 0; r < 8; ++r)
            out[(size_t)(row0 + r + 8 * lg) * COUT + g * DGRP + nt * 16 + ln] = c[r];
    }
}

extern "C" void kernel_launch(void* const* d_in, const int* in_sizes, int n_in,
                              void* d_out, int out_size, void* d_ws, size_t ws_size,
                              hipStream_t stream) {
    (void)in_sizes; (void)n_in; (void)out_size; (void)ws_size;
    const float* x   = (const float*)d_in[0];
    const float* Wp  = (const float*)d_in[1];
    const float* bp  = (const float*)d_in[2];
    const float* cb  = (const float*)d_in[3];
    const float* gum = (const float*)d_in[4];
    float* out = (float*)d_out;

    _Float16* wpf = (_Float16*)d_ws;                 // 640*512  halves = 640 KB
    _Float16* cbf = wpf + (size_t)NE * CIN;          // 2*320*256 halves = 320 KB

    swizzle_wp<<<(NE * CIN + 255) / 256, 256, 0, stream>>>(Wp, wpf);
    swizzle_cb<<<(NGRP * ENTRY * DGRP + 255) / 256, 256, 0, stream>>>(cb, cbf);
    quantizer_fused<<<BT / 16, 128, 0, stream>>>(x, bp, gum, wpf, cbf, out);
}
